// Discriminator_32598801777318
// MI455X (gfx1250) — matrix operand order repaired and checked by reading, not compiled, not run on hardware
//
#include <hip/hip_runtime.h>

// ============================================================================
// CDNA5 (gfx1250) windowed-attention discriminator.
//
// * All heavy math via v_wmma_f32_16x16x32_bf16 (bf16 operands, f32 accum).
// * Tile staging via GLOBAL_LOAD_ASYNC_TO_LDS_B128 (ASYNCcnt-tracked),
//   double-buffered; plain B128 load/store fallback if builtin is absent.
// * Activations flow between GEMMs as bf16 (f32 kept only where needed as
//   residuals), eliminating separate conversion passes.
// * Token-major windowed layout [B*64win*64tok, C]: every conv1x1/qkv/proj
//   is a GEMM; GroupNorm is layout-agnostic; attention is per-(window,head)
//   in LDS.
// ============================================================================

typedef __attribute__((ext_vector_type(16))) __bf16 v16bf;
typedef __attribute__((ext_vector_type(8)))  float  v8f;
typedef __attribute__((ext_vector_type(4)))  int    v4i;

union ABf { v16bf v; unsigned int u[8]; unsigned short h[16]; };

struct __align__(16) Chunk16 { unsigned int x[4]; };

#define AS1 __attribute__((address_space(1)))
#define AS3 __attribute__((address_space(3)))

#if defined(__has_builtin)
#if __has_builtin(__builtin_amdgcn_global_load_async_to_lds_b128)
#define USE_ASYNC_LDS 1
#endif
#endif

__device__ __forceinline__ void copy16(void* lds, const void* g) {
#ifdef USE_ASYNC_LDS
  __builtin_amdgcn_global_load_async_to_lds_b128(
      (AS1 v4i*)(unsigned long long)(uintptr_t)g,
      (AS3 v4i*)(unsigned int)(uintptr_t)lds, 0, 0);
#else
  *(Chunk16*)lds = *(const Chunk16*)g;
#endif
}

template <int N> __device__ __forceinline__ void async_wait() {
#ifdef USE_ASYNC_LDS
#if __has_builtin(__builtin_amdgcn_s_wait_asynccnt)
  __builtin_amdgcn_s_wait_asynccnt(N);
#else
  asm volatile("s_wait_asynccnt %0" ::"i"(N) : "memory");
#endif
#endif
}

__device__ __forceinline__ unsigned short f2bf(float f) {
  unsigned int x = __float_as_uint(f);
  unsigned int r = x + 0x7FFFu + ((x >> 16) & 1u);  // round-to-nearest-even
  return (unsigned short)(r >> 16);
}

// ---------------------------------------------------------------------------
// f32 -> bf16 (weights only; activations convert inline in GEMM/GN/attn)
// ---------------------------------------------------------------------------
__global__ __launch_bounds__(256) void cvt_kernel(const float* __restrict__ in,
                                                  unsigned short* __restrict__ out,
                                                  size_t n) {
  size_t i = (size_t)blockIdx.x * 256 + threadIdx.x;
  if (i < n) out[i] = f2bf(in[i]);
}

// ---------------------------------------------------------------------------
// Patch-embed conv 4x4/s4 (3->192) + bias, windowed token layout (f32 out).
// ---------------------------------------------------------------------------
__global__ __launch_bounds__(256) void embed_kernel(const float* __restrict__ img,
                                                    const float* __restrict__ w,
                                                    const float* __restrict__ bias,
                                                    float* __restrict__ out) {
  size_t gid = (size_t)blockIdx.x * 256 + threadIdx.x;
  if (gid >= (size_t)65536 * 192) return;
  int o = (int)(gid % 192);
  size_t t = gid / 192;
  int b = (int)(t >> 12);
  int r = (int)(t & 4095);
  int win = r >> 6, tok = r & 63;
  int hh = ((win >> 3) << 3) + (tok >> 3);
  int ww = ((win & 7) << 3) + (tok & 7);
  float acc = bias[o];
  #pragma unroll
  for (int ci = 0; ci < 3; ci++)
    #pragma unroll
    for (int py = 0; py < 4; py++)
      #pragma unroll
      for (int px = 0; px < 4; px++)
        acc += img[(((size_t)b * 3 + ci) * 256 + hh * 4 + py) * 256 + ww * 4 + px]
             * w[((o * 3 + ci) * 4 + py) * 4 + px];
  out[t * 192 + o] = acc;
}

// ---------------------------------------------------------------------------
// bf16 WMMA GEMM:  Y[M,N] = A[M,K] @ Bt[N,K]^T + bias
//   outputs: Cf (f32, nullable) and/or Cbf (bf16, nullable)
// Requirements (all call sites satisfy): M%128==0, N%64==0, K%32==0.
// Block: 256 threads = 8 wave32s; tile 128x64, K-step 32, double-buffered
// async global->LDS staging; each wave does a 32x32 tile (2x2 WMMA frags).
// ---------------------------------------------------------------------------
#define GBM 128
#define GBN 64
#define GBK 32
#define GLD 40  // LDS row pitch in bf16 (80B: 16B-aligned, conflict-free)

__global__ __launch_bounds__(256) void gemm_bf16(const unsigned short* __restrict__ A,
                                                 const unsigned short* __restrict__ Bt,
                                                 const float* __restrict__ bias,
                                                 float* __restrict__ Cf,
                                                 unsigned short* __restrict__ Cbf,
                                                 int M, int N, int K) {
  __shared__ __align__(16) unsigned short sA[2][GBM * GLD];
  __shared__ __align__(16) unsigned short sB[2][GBN * GLD];
  int tid = threadIdx.x;
  int wave = tid >> 5, lane = tid & 31;
  int bm = blockIdx.x * GBM, bn = blockIdx.y * GBN;
  int wm = (wave & 3) * 32, wn = (wave >> 2) * 32;
  int mrow = lane & 15, khalf = (lane >> 4) & 1;

  auto stage = [&](int buf, int k0) {
    // A tile 128x32 bf16 = 512 16B chunks, 2 per thread
    #pragma unroll
    for (int s = 0; s < 2; s++) {
      int c = tid + s * 256;
      int r = c >> 2, c8 = (c & 3) * 8;
      copy16(&sA[buf][r * GLD + c8], A + (size_t)(bm + r) * K + k0 + c8);
    }
    // B tile 64x32 bf16 = 256 chunks, 1 per thread
    {
      int r = tid >> 2, c8 = (tid & 3) * 8;
      copy16(&sB[buf][r * GLD + c8], Bt + (size_t)(bn + r) * K + k0 + c8);
    }
  };

  v8f acc[2][2] = {};
  const int nk = K / GBK;
  stage(0, 0);
  for (int i = 0; i < nk; i++) {
    int buf = i & 1;
    if (i + 1 < nk) {
      stage(buf ^ 1, (i + 1) * GBK);
      async_wait<3>();  // allow the 3 just-issued per-wave loads to remain
    } else {
      async_wait<0>();
    }
    __syncthreads();

    ABf af[2], bf[2];
    #pragma unroll
    for (int x = 0; x < 2; x++) {
      const unsigned short* ap = &sA[buf][(wm + x * 16 + mrow) * GLD];
      #pragma unroll
      for (int e2 = 0; e2 < 8; e2++) {
        int kk = e2 * 2 + khalf * 16;                                // contiguous k (matches B)
        af[x].u[e2] = *(const unsigned int*)(ap + kk);
      }
      const unsigned short* bp = &sB[buf][(wn + x * 16 + mrow) * GLD];
      #pragma unroll
      for (int e2 = 0; e2 < 8; e2++) {
        int kk = e2 * 2 + khalf * 16;                                // contiguous k (matches A)
        bf[x].u[e2] = *(const unsigned int*)(bp + kk);
      }
    }
    #pragma unroll
    for (int x = 0; x < 2; x++)
      #pragma unroll
      for (int y = 0; y < 2; y++)
        acc[x][y] = __builtin_amdgcn_wmma_f32_16x16x32_bf16(
            false, af[x].v, false, bf[y].v, (short)0, acc[x][y], false, false);
    __syncthreads();
  }

  // C/D layout: N = lane%16, M = vgpr + 8*(lane>=16)
  int cn = lane & 15, mh = (lane >> 4) * 8;
  #pragma unroll
  for (int x = 0; x < 2; x++)
    #pragma unroll
    for (int y = 0; y < 2; y++) {
      int gn = bn + wn + y * 16 + cn;
      float bv = bias ? bias[gn] : 0.0f;
      #pragma unroll
      for (int v = 0; v < 8; v++) {
        int gm = bm + wm + x * 16 + mh + v;
        float val = acc[x][y][v] + bv;
        size_t idx = (size_t)gm * N + gn;
        if (Cf) Cf[idx] = val;
        if (Cbf) Cbf[idx] = f2bf(val);
      }
    }
}

// ---------------------------------------------------------------------------
// Windowed MHA, one workgroup (4 wave32s) per (window, head), bf16 in/out.
// Qb[M,192]; KVb[M,384] (k: cols 0..191, v: 192..383); Outb[M,192].
// Outb may alias Qb (each workgroup writes exactly the slice it read).
// S = QK^T*scale + relbias -> softmax -> O = PV, WMMA throughout.
// ---------------------------------------------------------------------------
__global__ __launch_bounds__(128) void attn_kernel(const unsigned short* __restrict__ Qb,
                                                   const unsigned short* __restrict__ KVb,
                                                   const float* __restrict__ relb,
                                                   unsigned short* __restrict__ Outb) {
  __shared__ __align__(16) unsigned short sQ[64 * GLD];
  __shared__ __align__(16) unsigned short sK[64 * GLD];
  __shared__ __align__(16) unsigned short sV[64 * GLD];
  __shared__ __align__(16) float          sS[64 * 66];
  __shared__ __align__(16) unsigned short sP[64 * 66];

  int head = blockIdx.x % 6;
  int win  = blockIdx.x / 6;
  size_t base = (size_t)win * 64;
  int tid = threadIdx.x, lane = tid & 31, wave = tid >> 5;
  int mrow = lane & 15, khalf = (lane >> 4) & 1;

  // stage Q,K,V 64x32 bf16 (64B per row = 4 chunks; 256 chunks each)
  #pragma unroll
  for (int s = 0; s < 2; s++) {
    int c = tid + s * 128;
    int t = c >> 2, c8 = (c & 3) * 8;
    copy16(&sQ[t * GLD + c8], Qb  + (base + t) * 192 + head * 32 + c8);
    copy16(&sK[t * GLD + c8], KVb + (base + t) * 384 + head * 32 + c8);
    copy16(&sV[t * GLD + c8], KVb + (base + t) * 384 + 192 + head * 32 + c8);
  }
  async_wait<0>();
  __syncthreads();

  // S = Q @ K^T : each wave a 16-row strip, 4 N-tiles, K=32 (one WMMA each)
  {
    ABf a, b;
    const unsigned short* ap = &sQ[(wave * 16 + mrow) * GLD];
    #pragma unroll
    for (int e2 = 0; e2 < 8; e2++)
      a.u[e2] = *(const unsigned int*)(ap + (e2 * 2 + khalf * 16));
    #pragma unroll
    for (int nt = 0; nt < 4; nt++) {
      const unsigned short* bp = &sK[(nt * 16 + mrow) * GLD];
      #pragma unroll
      for (int e2 = 0; e2 < 8; e2++)
        b.u[e2] = *(const unsigned int*)(bp + (e2 * 2 + khalf * 16));
      v8f s_acc = {};
      s_acc = __builtin_amdgcn_wmma_f32_16x16x32_bf16(false, a.v, false, b.v, (short)0,
                                                      s_acc, false, false);
      int cn = lane & 15, mh = (lane >> 4) * 8;
      #pragma unroll
      for (int v = 0; v < 8; v++)
        sS[(wave * 16 + mh + v) * 66 + nt * 16 + cn] = s_acc[v];
    }
  }
  __syncthreads();

  // softmax rows (threads 0..63); scale = 1/sqrt(32); arithmetic rel-bias
  if (tid < 64) {
    const float scale = 0.17677669529663687f;
    int i = tid, yi = i >> 3, xi = i & 7;
    float m = -1e30f;
    for (int j = 0; j < 64; j++) {
      int idx = (yi - (j >> 3) + 7) * 15 + (xi - (j & 7) + 7);
      m = fmaxf(m, sS[i * 66 + j] * scale + relb[idx * 6 + head]);
    }
    float sum = 0.0f;
    for (int j = 0; j < 64; j++) {
      int idx = (yi - (j >> 3) + 7) * 15 + (xi - (j & 7) + 7);
      sum += __expf(sS[i * 66 + j] * scale + relb[idx * 6 + head] - m);
    }
    float inv = 1.0f / sum;
    for (int j = 0; j < 64; j++) {
      int idx = (yi - (j >> 3) + 7) * 15 + (xi - (j & 7) + 7);
      sP[i * 66 + j] = f2bf(__expf(sS[i * 66 + j] * scale + relb[idx * 6 + head] - m) * inv);
    }
  }
  __syncthreads();

  // O = P @ V : each wave a 16-row strip, 2 N-tiles (head dim 32), K = 2x32
  {
    #pragma unroll
    for (int nt = 0; nt < 2; nt++) {
      v8f o_acc = {};
      #pragma unroll
      for (int kt = 0; kt < 2; kt++) {
        ABf a, b;
        const unsigned short* ap = &sP[(wave * 16 + mrow) * 66 + kt * 32];
        #pragma unroll
        for (int e2 = 0; e2 < 8; e2++)
          a.u[e2] = *(const unsigned int*)(ap + (e2 * 2 + khalf * 16));
        #pragma unroll
        for (int e2 = 0; e2 < 8; e2++) {
          int k0 = e2 * 2 + khalf * 16;  // B(k,n) = V[kt*32+k][nt*16+n]
          unsigned int lo = sV[(kt * 32 + k0) * GLD + nt * 16 + mrow];
          unsigned int hi = sV[(kt * 32 + k0 + 1) * GLD + nt * 16 + mrow];
          b.u[e2] = lo | (hi << 16);
        }
        o_acc = __builtin_amdgcn_wmma_f32_16x16x32_bf16(false, a.v, false, b.v, (short)0,
                                                        o_acc, false, false);
      }
      int cn = lane & 15, mh = (lane >> 4) * 8;
      #pragma unroll
      for (int v = 0; v < 8; v++)
        Outb[(base + wave * 16 + mh + v) * 192 + head * 32 + nt * 16 + cn] = f2bf(o_acc[v]);
    }
  }
}

// ---------------------------------------------------------------------------
// Fused (optional residual) + GroupNorm + (optional ReLU); f32 and/or bf16
// outputs. One block per (batch, group); population var, eps=1e-5. In-place
// safe w.r.t. x/res vs outf.
// ---------------------------------------------------------------------------
__global__ __launch_bounds__(256) void groupnorm_kernel(const float* __restrict__ x,
                                                        const float* __restrict__ res,
                                                        const float* __restrict__ gamma,
                                                        const float* __restrict__ beta,
                                                        float* __restrict__ outf,
                                                        unsigned short* __restrict__ outbf,
                                                        int Cn, int cpg, int relu) {
  int G = Cn / cpg;
  int b = blockIdx.x / G, g = blockIdx.x % G;
  size_t tokBase = (size_t)b * 4096;
  int c0 = g * cpg;
  int n = 4096 * cpg;

  float s = 0.0f, s2 = 0.0f;
  for (int i = threadIdx.x; i < n; i += 256) {
    int t = i / cpg, c = c0 + i % cpg;
    float v = x[(tokBase + t) * Cn + c];
    if (res) v += res[(tokBase + t) * Cn + c];
    s += v; s2 += v * v;
  }
  __shared__ float rs[256], rs2[256];
  rs[threadIdx.x] = s; rs2[threadIdx.x] = s2;
  __syncthreads();
  for (int off = 128; off > 0; off >>= 1) {
    if (threadIdx.x < off) {
      rs[threadIdx.x]  += rs[threadIdx.x + off];
      rs2[threadIdx.x] += rs2[threadIdx.x + off];
    }
    __syncthreads();
  }
  float mean = rs[0] / (float)n;
  float var  = rs2[0] / (float)n - mean * mean;
  float rstd = rsqrtf(var + 1e-5f);

  for (int i = threadIdx.x; i < n; i += 256) {
    int t = i / cpg, c = c0 + i % cpg;
    size_t idx = (tokBase + t) * Cn + c;
    float v = x[idx];
    if (res) v += res[idx];
    v = (v - mean) * rstd * gamma[c] + beta[c];
    if (relu) v = fmaxf(v, 0.0f);
    if (outf)  outf[idx] = v;
    if (outbf) outbf[idx] = f2bf(v);
  }
}

// ---------------------------------------------------------------------------
// Final: h*||fw|| * fw/||fw|| == h_raw ⊙ fw  (logits and dir_ identical).
// ---------------------------------------------------------------------------
__global__ __launch_bounds__(256) void final_kernel(const float* __restrict__ q,
                                                    const float* __restrict__ fw,
                                                    float* __restrict__ out) {
  size_t gid = (size_t)blockIdx.x * 256 + threadIdx.x;
  if (gid >= (size_t)65536 * 192) return;
  int c = (int)(gid % 192);
  size_t t = gid / 192;
  int b = (int)(t >> 12);
  int r = (int)(t & 4095);
  int win = r >> 6, tok = r & 63;
  int hh = ((win >> 3) << 3) + (tok >> 3);
  int ww = ((win & 7) << 3) + (tok & 7);
  size_t d = (size_t)c * 4096 + hh * 64 + ww;
  float v = q[t * 192 + c] * fw[d];
  const size_t D = 786432;
  out[(size_t)b * D + d] = v;                   // logits
  out[(size_t)16 * D + (size_t)b * D + d] = v;  // dir_
}

// ===========================================================================
// Host orchestration
// ===========================================================================

// Input leaf indices (setup_inputs dict, insertion order, recursive)
enum {
  IN_IMGS = 0, IN_EDGES = 1,
  E_W = 2, E_B = 3, E_GG = 4, E_GB = 5,
  ENC0 = 6,   // +0 qkv_w +1 qkv_b +2 proj_w +3 proj_b +4 rel_bias
              // +5 n1_g +6 n1_b +7 n2_g +8 n2_b
              // +9 w1 +10 b1 +11 gn_g +12 gn_b +13 w2 +14 b2      (15 leaves)
  ENC1 = 21,
  DEC0 = 36,  // +0..4 sattn, +5..9 cattn, +10 n1_g +11 n1_b +12 n2_g +13 n2_b
              // +14 n3_g +15 n3_b, +16 w1 +17 b1 +18 gn_g +19 gn_b +20 w2 +21 b2
  DEC1 = 58,
  Q_W = 80, Q_B = 81, FC_W = 82
};

extern "C" void kernel_launch(void* const* d_in, const int* in_sizes, int n_in,
                              void* d_out, int out_size, void* d_ws, size_t ws_size,
                              hipStream_t stream) {
  (void)in_sizes; (void)n_in; (void)out_size; (void)ws_size;
  const size_t MT = 65536;  // B * 64 windows * 64 tokens

#define P(i) ((const float*)d_in[(i)])

  char* wsb = (char*)d_ws;
  size_t off = 0;
  auto take = [&](size_t bytes) -> char* {
    char* p = wsb + off;
    off += (bytes + 255) & ~(size_t)255;
    return p;
  };
  // f32 buffers
  float* ctx_f = (float*)take(MT * 192 * 4);
  float* h_f   = (float*)take(MT * 192 * 4);
  float* t0    = (float*)take(MT * 768 * 4);  // mlp hidden (f32, pre-GN)
  float* t1    = (float*)take(MT * 192 * 4);  // gemm f32 out / embed out
  float* t2_f  = (float*)take(MT * 192 * 4);  // enc h
  float* t3_f  = (float*)take(MT * 192 * 4);  // dec h'
  // bf16 buffers
  unsigned short* ctx_b = (unsigned short*)take(MT * 192 * 2);
  unsigned short* h_b   = (unsigned short*)take(MT * 192 * 2);
  unsigned short* qa_b  = (unsigned short*)take(MT * 192 * 2);  // q / attn-out
  unsigned short* kv_b  = (unsigned short*)take(MT * 384 * 2);
  unsigned short* t2_b  = (unsigned short*)take(MT * 192 * 2);  // enc h / dec z
  unsigned short* t3_b  = (unsigned short*)take(MT * 192 * 2);  // dec h'
  unsigned short* xb768 = (unsigned short*)take(MT * 768 * 2);  // gn(hidden)
  unsigned short* wbf   = (unsigned short*)take((size_t)768 * 768 * 2);

  auto gemm = [&](const unsigned short* xb, const float* w, const float* bias,
                  float* yf, unsigned short* ybf, int N, int K) {
    size_t nw = (size_t)N * K;
    cvt_kernel<<<dim3((unsigned)((nw + 255) / 256)), 256, 0, stream>>>(w, wbf, nw);
    gemm_bf16<<<dim3((unsigned)(MT / GBM), (unsigned)(N / GBN)), 256, 0, stream>>>(
        xb, wbf, bias, yf, ybf, (int)MT, N, K);
  };
  auto gnorm = [&](const float* x, const float* res, const float* g, const float* b,
                   float* outf, unsigned short* outbf, int Cn, int relu) {
    groupnorm_kernel<<<dim3(16 * 32), 256, 0, stream>>>(x, res, g, b, outf, outbf,
                                                        Cn, Cn / 32, relu);
  };
  auto attn = [&](const unsigned short* q, const unsigned short* kv, const float* relb,
                  unsigned short* outb) {
    attn_kernel<<<dim3(16 * 64 * 6), 128, 0, stream>>>(q, kv, relb, outb);
  };

  const unsigned embGrid = (unsigned)((MT * 192 + 255) / 256);

  // ---- embed(imgs) -> ctx ; embed(edges) -> h ----
  embed_kernel<<<embGrid, 256, 0, stream>>>(P(IN_IMGS), P(E_W), P(E_B), t1);
  gnorm(t1, nullptr, P(E_GG), P(E_GB), ctx_f, ctx_b, 192, 1);
  embed_kernel<<<embGrid, 256, 0, stream>>>(P(IN_EDGES), P(E_W), P(E_B), t1);
  gnorm(t1, nullptr, P(E_GG), P(E_GB), h_f, h_b, 192, 1);

  for (int blk = 0; blk < 2; blk++) {
    const int eb = (blk == 0) ? ENC0 : ENC1;
    const int db = (blk == 0) ? DEC0 : DEC1;

    // ======== encoder_block(ctx) ========
    gemm(ctx_b, P(eb + 0),           P(eb + 1),       nullptr, qa_b, 192, 192);  // q
    gemm(ctx_b, P(eb + 0) + 192*192, P(eb + 1) + 192, nullptr, kv_b, 384, 192);  // k|v
    attn(qa_b, kv_b, P(eb + 4), qa_b);
    gemm(qa_b, P(eb + 2), P(eb + 3), t1, nullptr, 192, 192);                     // proj
    gnorm(ctx_f, t1, P(eb + 5), P(eb + 6), t2_f, t2_b, 192, 1);                  // h
    gemm(t2_b, P(eb + 9), P(eb + 10), t0, nullptr, 768, 192);                    // mlp hid
    gnorm(t0, nullptr, P(eb + 11), P(eb + 12), nullptr, xb768, 768, 1);
    gemm(xb768, P(eb + 13), P(eb + 14), t1, nullptr, 192, 768);                  // mlp out
    gnorm(t2_f, t1, P(eb + 7), P(eb + 8), ctx_f, ctx_b, 192, 1);                 // new ctx

    // ======== decoder_block(h, ctx) ========
    gemm(h_b, P(db + 0),           P(db + 1),       nullptr, qa_b, 192, 192);    // q
    gemm(h_b, P(db + 0) + 192*192, P(db + 1) + 192, nullptr, kv_b, 384, 192);    // k|v
    attn(qa_b, kv_b, P(db + 4), qa_b);
    gemm(qa_b, P(db + 2), P(db + 3), t1, nullptr, 192, 192);                     // s proj
    gnorm(h_f, t1, P(db + 10), P(db + 11), t3_f, t3_b, 192, 1);                  // h'
    gemm(t3_b,  P(db + 5),           P(db + 6),       nullptr, qa_b, 192, 192);  // q(h')
    gemm(ctx_b, P(db + 5) + 192*192, P(db + 6) + 192, nullptr, kv_b, 384, 192);  // kv(ctx)
    attn(qa_b, kv_b, P(db + 9), qa_b);
    gemm(qa_b, P(db + 7), P(db + 8), t1, nullptr, 192, 192);                     // c proj
    gnorm(t3_f, t1, P(db + 12), P(db + 13), nullptr, t2_b, 192, 1);              // z (bf)
    gemm(t2_b, P(db + 16), P(db + 17), t0, nullptr, 768, 192);                   // mlp hid
    gnorm(t0, nullptr, P(db + 18), P(db + 19), nullptr, xb768, 768, 1);
    gemm(xb768, P(db + 20), P(db + 21), t1, nullptr, 192, 768);                  // mlp out
    gnorm(t3_f, t1, P(db + 14), P(db + 15), h_f, h_b, 192, 1);                   // new h
  }

  // ---- quant conv + fc direction products ----
  gemm(h_b, P(Q_W), P(Q_B), t1, nullptr, 192, 192);
  final_kernel<<<embGrid, 256, 0, stream>>>(t1, P(FC_W), (float*)d_out);

#undef P
}
